// SparseInvertor_66314295050800
// MI455X (gfx1250) — matrix operand order repaired and checked by reading, not compiled, not run on hardware
//
#include <hip/hip_runtime.h>
#include <math.h>

#define T_   4096
#define D_   256
#define H_   1024
#define E_   8
#define K_   2
#define C_   1280
#define CT_  (C_/32)   // 40 row-tiles per expert

typedef __attribute__((ext_vector_type(16))) __bf16 v16bf;
typedef __attribute__((ext_vector_type(8)))  float  v8f;

union FragB {
    v16bf v;
    uint4 q[2];
    unsigned short u[16];
};

static __device__ __forceinline__ unsigned short f2bf(float f) {
    unsigned int u = __float_as_uint(f);
    unsigned int r = (u + 0x7FFFu + ((u >> 16) & 1u)) >> 16;  // RNE
    return (unsigned short)r;
}

static __device__ __forceinline__ float gelu_tanh(float x) {
    float x3 = x * x * x;
    float inner = 0.7978845608028654f * (x + 0.044715f * x3);
    return 0.5f * x * (1.0f + tanhf(inner));
}

// ---------------- gating: softmax(x@Wg), top-2, renorm ----------------
__global__ void gate_topk_kernel(const float* __restrict__ x, const float* __restrict__ Wg,
                                 float* __restrict__ gates, int* __restrict__ eidx,
                                 float* __restrict__ ew) {
    const int wave = threadIdx.x >> 5, lane = threadIdx.x & 31;
    const int t = blockIdx.x * 8 + wave;
    float p[E_];
#pragma unroll
    for (int e = 0; e < E_; e++) p[e] = 0.f;
    for (int d = lane; d < D_; d += 32) {
        float xv = x[(size_t)t * D_ + d];
        const float* wr = &Wg[d * E_];
#pragma unroll
        for (int e = 0; e < E_; e++) p[e] += xv * wr[e];
    }
#pragma unroll
    for (int off = 16; off > 0; off >>= 1)
#pragma unroll
        for (int e = 0; e < E_; e++) p[e] += __shfl_xor(p[e], off, 32);

    float m = p[0];
#pragma unroll
    for (int e = 1; e < E_; e++) m = fmaxf(m, p[e]);
    float g[E_], s = 0.f;
#pragma unroll
    for (int e = 0; e < E_; e++) { g[e] = __expf(p[e] - m); s += g[e]; }
    float inv = 1.f / s;
#pragma unroll
    for (int e = 0; e < E_; e++) g[e] *= inv;

    if (lane < E_) gates[(size_t)t * E_ + lane] = g[lane];
    if (lane == 0) {
        int e0 = 0; float g0 = g[0];
#pragma unroll
        for (int e = 1; e < E_; e++) if (g[e] > g0) { g0 = g[e]; e0 = e; }
        int e1 = -1; float g1 = -1.f;
#pragma unroll
        for (int e = 0; e < E_; e++) if (e != e0 && g[e] > g1) { g1 = g[e]; e1 = e; }
        float sn = 1.f / (g0 + g1);
        eidx[2 * t] = e0; eidx[2 * t + 1] = e1;
        ew[2 * t] = g0 * sn; ew[2 * t + 1] = g1 * sn;
    }
}

// ---------------- deterministic per-expert position scan ----------------
__global__ void scan_pos_kernel(const int* __restrict__ eidx, int* __restrict__ pos,
                                int* __restrict__ counts) {
    __shared__ int scnt[256][E_];
    const int tid = threadIdx.x;
    const int base = tid * 32;   // 8192 assignments / 256 threads
    int c[E_];
#pragma unroll
    for (int e = 0; e < E_; e++) c[e] = 0;
    for (int j = 0; j < 32; j++) c[eidx[base + j]]++;
#pragma unroll
    for (int e = 0; e < E_; e++) scnt[tid][e] = c[e];
    __syncthreads();
    if (tid < E_) {
        int run = 0;
        for (int i = 0; i < 256; i++) { int v = scnt[i][tid]; scnt[i][tid] = run; run += v; }
        counts[tid] = run;
    }
    __syncthreads();
    int off[E_];
#pragma unroll
    for (int e = 0; e < E_; e++) off[e] = scnt[tid][e];
    for (int j = 0; j < 32; j++) { int e = eidx[base + j]; pos[base + j] = off[e]++; }
}

// ---------------- load-balance loss ----------------
__global__ void lb_loss_kernel(const float* __restrict__ gates, const int* __restrict__ counts,
                               float* __restrict__ out) {
    const int wave = threadIdx.x >> 5, lane = threadIdx.x & 31;
    __shared__ float part[E_];
    float s = 0.f;
    for (int t = lane; t < T_; t += 32) s += gates[(size_t)t * E_ + wave];
#pragma unroll
    for (int off = 16; off > 0; off >>= 1) s += __shfl_xor(s, off, 32);
    if (lane == 0) {
        float pmean = s / (float)T_;
        float density = (float)counts[wave] / (float)(T_ * K_);
        part[wave] = density * pmean;
    }
    __syncthreads();
    if (threadIdx.x == 0) {
        float tot = 0.f;
        for (int e = 0; e < E_; e++) tot += part[e];
        *out = (float)E_ * tot;
    }
}

// ---------------- zero fill ----------------
__global__ void zero_u32_kernel(unsigned int* __restrict__ p, int n) {
    for (int i = blockIdx.x * blockDim.x + threadIdx.x; i < n; i += gridDim.x * blockDim.x)
        p[i] = 0u;
}

// ---------------- dispatch tokens -> bf16 capacity buffers ----------------
__global__ void dispatch_kernel(const float* __restrict__ x, const int* __restrict__ eidx,
                                const int* __restrict__ pos, unsigned short* __restrict__ bufbf) {
    const int wave = threadIdx.x >> 5, lane = threadIdx.x & 31;
    const int a = blockIdx.x * 8 + wave;   // 1024 blocks * 8 waves = 8192 assignments
    const int e = eidx[a], p = pos[a];
    if (p < C_) {
        const int t = a >> 1;
        const size_t dst = (size_t)(e * C_ + p) * D_;
        for (int d = lane; d < D_; d += 32)
            bufbf[dst + d] = f2bf(x[(size_t)t * D_ + d]);
    }
}

// ---------------- pack W (fp32 [E,Kdim,N]) into bf16 WMMA-B tiles ----------------
// Tile = 32(K) x 16(N). Lane l holds col n=l%16, h=l/16, element j: K = 8h + j (j<8), 16 + 8h + (j-8) (j>=8),
// stored contiguously: tile_base + l*16 + j  -> consumer does two 16B loads/lane.
__global__ void pack_b_kernel(const float* __restrict__ src, unsigned short* __restrict__ dst,
                              int Kdim, int N, int total) {
    const int tilesN = N >> 4;
    const int tilesPerExpert = (Kdim >> 5) * tilesN;
    for (int s = blockIdx.x * blockDim.x + threadIdx.x; s < total; s += gridDim.x * blockDim.x) {
        int e = s / (Kdim * N);
        int r = s - e * (Kdim * N);
        int k = r / N;
        int n = r - k * N;
        int tile = (k >> 5) * tilesN + (n >> 4);
        int l = ((k >> 3) & 1) * 16 + (n & 15);
        int j = (k & 7) + ((k >> 4) & 1) * 8;
        size_t d = ((size_t)(e * tilesPerExpert + tile)) * 512 + (size_t)l * 16 + j;
        dst[d] = f2bf(src[s]);
    }
}

// ---------------- fused expert MLP: y = gelu(x@W1+b1)@W2+b2 ----------------
// grid = E_*CT_ blocks, 256 threads (8 waves). Block owns a 32x256 output tile.
__global__ void moe_mlp_kernel(const unsigned short* __restrict__ bufbf,
                               const unsigned short* __restrict__ W1bf,
                               const float* __restrict__ b1,
                               const unsigned short* __restrict__ W2bf,
                               const float* __restrict__ b2,
                               float* __restrict__ yb) {
    __shared__ __align__(16) unsigned short xs[32 * 256];  // 16KB bf16 input tile
    __shared__ __align__(16) unsigned short hs[32 * 64];   // 4KB bf16 hidden chunk
    const int tid = threadIdx.x, lane = tid & 31, wave = tid >> 5;
    const int e = blockIdx.x / CT_, ctile = blockIdx.x % CT_;

    {   // cooperative load of the 32x256 bf16 tile (rows contiguous in bufbf)
        const uint4* src = (const uint4*)(bufbf + (size_t)(e * C_ + ctile * 32) * D_);
        uint4* dst = (uint4*)xs;
#pragma unroll
        for (int i = 0; i < 4; i++) dst[tid + i * 256] = src[tid + i * 256];
    }
    __syncthreads();

    v8f c00 = {}, c01 = {}, c10 = {}, c11 = {};
    const int mi = wave >> 2, nj = wave & 3;       // h-phase: this wave's 16x16 h tile
    const int nw = wave * 32;                      // y-phase: this wave's 32 output cols
    const int koff = (lane >> 4) * 8;              // A-fragment K sub-offset (ISA layout)
    const int ncol = lane & 15;
    const int rA = mi * 16 + (lane & 15);          // h-phase A row in xs
    const size_t wbase = (size_t)e * 512 * 512;    // 512 tiles/expert * 512 elems/tile

    for (int hblk = 0; hblk < H_; hblk += 64) {
        // ---- h tile: [32 x 64] columns [hblk, hblk+64), one 16x16 per wave ----
        v8f acc = {};
        const unsigned short* xr = xs + rA * 256;
#pragma unroll
        for (int kb = 0; kb < D_; kb += 32) {
            FragB fa, fb;
            fa.q[0] = *(const uint4*)(xr + kb + koff);
            fa.q[1] = *(const uint4*)(xr + kb + 16 + koff);
            const int tile = (kb >> 5) * (H_ >> 4) + (hblk >> 4) + nj;
            const uint4* bp = (const uint4*)(W1bf + wbase + (size_t)tile * 512 + lane * 16);
            fb.q[0] = bp[0]; fb.q[1] = bp[1];
            acc = __builtin_amdgcn_wmma_f32_16x16x32_bf16(false, fa.v, false, fb.v,
                                                          (short)0, acc, false, false);
        }
        const float b1v = b1[e * H_ + hblk + nj * 16 + ncol];
        const int mbase = mi * 16 + koff;  // C layout: VGPR i -> M = i (+8 for hi lanes)
#pragma unroll
        for (int i = 0; i < 8; i++) {
            float hv = gelu_tanh(acc[i] + b1v);
            hs[(mbase + i) * 64 + nj * 16 + ncol] = f2bf(hv);
        }
        __syncthreads();

        // ---- y += h @ W2 for K in [hblk, hblk+64), this wave's cols [nw, nw+32) ----
        const unsigned short* h0r = hs + (lane & 15) * 64;
        const unsigned short* h1r = hs + (16 + (lane & 15)) * 64;
#pragma unroll
        for (int kk = 0; kk < 64; kk += 32) {
            FragB a0, a1, bb0, bb1;
            a0.q[0] = *(const uint4*)(h0r + kk + koff);
            a0.q[1] = *(const uint4*)(h0r + kk + 16 + koff);
            a1.q[0] = *(const uint4*)(h1r + kk + koff);
            a1.q[1] = *(const uint4*)(h1r + kk + 16 + koff);
            const int tile0 = ((hblk + kk) >> 5) * (D_ >> 4) + (nw >> 4);
            const uint4* b0p = (const uint4*)(W2bf + wbase + (size_t)tile0 * 512 + lane * 16);
            const uint4* b1p = (const uint4*)(W2bf + wbase + (size_t)(tile0 + 1) * 512 + lane * 16);
            bb0.q[0] = b0p[0]; bb0.q[1] = b0p[1];
            bb1.q[0] = b1p[0]; bb1.q[1] = b1p[1];
            c00 = __builtin_amdgcn_wmma_f32_16x16x32_bf16(false, a0.v, false, bb0.v, (short)0, c00, false, false);
            c01 = __builtin_amdgcn_wmma_f32_16x16x32_bf16(false, a0.v, false, bb1.v, (short)0, c01, false, false);
            c10 = __builtin_amdgcn_wmma_f32_16x16x32_bf16(false, a1.v, false, bb0.v, (short)0, c10, false, false);
            c11 = __builtin_amdgcn_wmma_f32_16x16x32_bf16(false, a1.v, false, bb1.v, (short)0, c11, false, false);
        }
        __syncthreads();
    }

    // ---- epilogue: + b2, write fp32 expert outputs ----
    const float b2v0 = b2[e * D_ + nw + ncol];
    const float b2v1 = b2[e * D_ + nw + 16 + ncol];
    const size_t rowbase = (size_t)(e * C_ + ctile * 32);
#pragma unroll
    for (int i = 0; i < 8; i++) {
        const int r0 = koff + i;       // rows 0..15 sub-block
        const int r1 = 16 + koff + i;  // rows 16..31 sub-block
        yb[(rowbase + r0) * D_ + nw + ncol]      = c00[i] + b2v0;
        yb[(rowbase + r0) * D_ + nw + 16 + ncol] = c01[i] + b2v1;
        yb[(rowbase + r1) * D_ + nw + ncol]      = c10[i] + b2v0;
        yb[(rowbase + r1) * D_ + nw + 16 + ncol] = c11[i] + b2v1;
    }
}

// ---------------- combine top-2 + fused L2 normalize ----------------
__global__ void combine_norm_kernel(const float* __restrict__ yb, const int* __restrict__ eidx,
                                    const float* __restrict__ ew, const int* __restrict__ pos,
                                    float* __restrict__ out) {
    const int t = blockIdx.x, d = threadIdx.x;
    const int a0 = 2 * t, a1 = 2 * t + 1;
    float v = 0.f;
    const int p0 = pos[a0];
    if (p0 < C_) v += ew[a0] * yb[(size_t)(eidx[a0] * C_ + p0) * D_ + d];
    const int p1 = pos[a1];
    if (p1 < C_) v += ew[a1] * yb[(size_t)(eidx[a1] * C_ + p1) * D_ + d];

    __shared__ float red[256];
    red[d] = v * v;
    __syncthreads();
    for (int s = 128; s > 0; s >>= 1) { if (d < s) red[d] += red[d + s]; __syncthreads(); }
    const float scale = 1.f / fmaxf(sqrtf(red[0]), 1e-12f);
    out[(size_t)t * D_ + d] = v * scale;
}

extern "C" void kernel_launch(void* const* d_in, const int* in_sizes, int n_in,
                              void* d_out, int out_size, void* d_ws, size_t ws_size,
                              hipStream_t stream) {
    (void)in_sizes; (void)n_in; (void)out_size; (void)ws_size;
    const float* y    = (const float*)d_in[0];
    const float* Wg0  = (const float*)d_in[1];
    const float* W1_0 = (const float*)d_in[2];
    const float* b1_0 = (const float*)d_in[3];
    const float* W2_0 = (const float*)d_in[4];
    const float* b2_0 = (const float*)d_in[5];
    const float* Wg1  = (const float*)d_in[6];
    const float* W1_1 = (const float*)d_in[7];
    const float* b1_1 = (const float*)d_in[8];
    const float* W2_1 = (const float*)d_in[9];
    const float* b2_1 = (const float*)d_in[10];
    float* out = (float*)d_out;

    // workspace carve-up (all chunks 256B-aligned sizes)
    char* w = (char*)d_ws;
    float* gates = (float*)w;            w += (size_t)T_ * E_ * 4;        // 128 KB
    int*   eidx  = (int*)w;              w += (size_t)T_ * K_ * 4;        // 32 KB
    float* ew    = (float*)w;            w += (size_t)T_ * K_ * 4;        // 32 KB
    int*   pos   = (int*)w;              w += (size_t)T_ * K_ * 4;        // 32 KB
    int*   counts= (int*)w;              w += 256;
    unsigned short* bufbf = (unsigned short*)w; w += (size_t)E_ * C_ * D_ * 2;  // 5 MB
    unsigned short* W1bf  = (unsigned short*)w; w += (size_t)E_ * D_ * H_ * 2;  // 4 MB
    unsigned short* W2bf  = (unsigned short*)w; w += (size_t)E_ * H_ * D_ * 2;  // 4 MB
    float* ybuf = (float*)w;             w += (size_t)E_ * C_ * D_ * 4;   // 10 MB
    float* h0   = (float*)w;             w += (size_t)T_ * D_ * 4;        // 4 MB

    const int nZero = E_ * C_ * D_ / 2;  // bf16 buffer as u32 words

    struct LayerArgs { const float *xin, *Wg, *W1, *b1, *W2, *b2; float *dst, *lb; };
    LayerArgs L[2] = {
        { y,  Wg0, W1_0, b1_0, W2_0, b2_0, h0,  out + (size_t)T_ * D_ + 0 },
        { h0, Wg1, W1_1, b1_1, W2_1, b2_1, out, out + (size_t)T_ * D_ + 1 },
    };

    for (int li = 0; li < 2; li++) {
        LayerArgs& A = L[li];
        gate_topk_kernel<<<T_ / 8, 256, 0, stream>>>(A.xin, A.Wg, gates, eidx, ew);
        scan_pos_kernel<<<1, 256, 0, stream>>>(eidx, pos, counts);
        lb_loss_kernel<<<1, 256, 0, stream>>>(gates, counts, A.lb);
        zero_u32_kernel<<<2560, 256, 0, stream>>>((unsigned int*)bufbf, nZero);
        dispatch_kernel<<<(T_ * K_) / 8, 256, 0, stream>>>(A.xin, eidx, pos, bufbf);
        pack_b_kernel<<<2048, 256, 0, stream>>>(A.W1, W1bf, D_, H_, E_ * D_ * H_);
        pack_b_kernel<<<2048, 256, 0, stream>>>(A.W2, W2bf, H_, D_, E_ * H_ * D_);
        moe_mlp_kernel<<<E_ * CT_, 256, 0, stream>>>(bufbf, W1bf, A.b1, W2bf, A.b2, ybuf);
        combine_norm_kernel<<<T_, 256, 0, stream>>>(ybuf, eidx, ew, pos, A.dst);
    }
}
